// MessagePassingEncoder_81217831568100
// MI455X (gfx1250) — matrix operand order repaired and checked by reading, not compiled, not run on hardware
//
#include <hip/hip_runtime.h>
#include <hip/hip_bf16.h>

// ---------- types ----------
typedef __attribute__((ext_vector_type(16))) __bf16 v16bf;
typedef __attribute__((ext_vector_type(8)))  float  v8f;
typedef __attribute__((ext_vector_type(4)))  float  v4f;
typedef __attribute__((ext_vector_type(4)))  unsigned int v4u;

union FragU {
    v4u   q[2];   // two 16-byte loads
    v16bf v;      // WMMA operand
};

#define GIN_D 96
#define GIN_L 3
#define TILE_ROWS 128
#define LDS_STRIDE 104   // multiple of 8 ushorts -> 16B-aligned fragment loads

static __device__ __forceinline__ unsigned short f2bf(float f) {
    unsigned int u = __float_as_uint(f);
    u += 0x7FFFu + ((u >> 16) & 1u);      // round-to-nearest-even
    return (unsigned short)(u >> 16);
}

// ---------- kernel 1: z = h (float4 copy) ----------
__global__ void gin_copy_f4(const float* __restrict__ in, float* __restrict__ out, int n4) {
    int t = blockIdx.x * blockDim.x + threadIdx.x;
    if (t < n4) ((v4f*)out)[t] = ((const v4f*)in)[t];
}

// ---------- kernel 2: z[dst] += h[src] over edges (8 channels / thread) ----------
__global__ void gin_scatter(const float* __restrict__ h, const int* __restrict__ src,
                            const int* __restrict__ dst, float* __restrict__ z, int E) {
    int t = blockIdx.x * blockDim.x + threadIdx.x;
    int tot = E * 12;                  // 96 channels / 8 per thread
    if (t >= tot) return;
    int e = t / 12;
    int g = (t % 12) * 8;
    int s = src[e], d = dst[e];
    v4f a = *(const v4f*)(h + (size_t)s * GIN_D + g);
    v4f b = *(const v4f*)(h + (size_t)s * GIN_D + g + 4);
    float* zp = z + (size_t)d * GIN_D + g;
    atomicAdd(zp + 0, a.x); atomicAdd(zp + 1, a.y);
    atomicAdd(zp + 2, a.z); atomicAdd(zp + 3, a.w);
    atomicAdd(zp + 4, b.x); atomicAdd(zp + 5, b.y);
    atomicAdd(zp + 6, b.z); atomicAdd(zp + 7, b.w);
}

// ---------- kernel 3: pre-swizzle weights to bf16 B-fragment layout; zero BN sums ----------
// flat index = ((ntile*3 + kchunk)*32 + lane)*16 + elem
//   n = ntile*16 + (lane&15),  k = kchunk*32 + (lane>>4)*16 + elem   (B is 32x16 K-major per tile)
__global__ void gin_prep(const float* __restrict__ W1, const float* __restrict__ W2,
                         unsigned short* __restrict__ w1s, unsigned short* __restrict__ w2s,
                         float* __restrict__ csum, float* __restrict__ csqs) {
    int t = blockIdx.x * blockDim.x + threadIdx.x;
    if (t < GIN_D)                 csum[t] = 0.f;
    else if (t < 2 * GIN_D)        csqs[t - GIN_D] = 0.f;
    const int PER = 6 * 3 * 32 * 16;   // 9216 per weight matrix
    if (t >= 2 * PER) return;
    int which = t / PER;
    int r = t - which * PER;
    int elem = r & 15;
    int lane = (r >> 4) & 31;
    int kc   = (r >> 9) % 3;
    int n    = r / 1536;
    int col = n * 16 + (lane & 15);
    int k   = kc * 32 + (lane >> 4) * 16 + elem;
    const float* W = which ? W2 : W1;
    unsigned short* o = which ? w2s : w1s;
    o[r] = f2bf(W[k * GIN_D + col]);
}

// ---------- kernel 4: fused GIN MLP  out = relu(z@W1+b1)@W2+b2  + BN partial sums ----------
__global__ __launch_bounds__(256)
void gin_mlp(const float* __restrict__ z,
             const unsigned short* __restrict__ w1s, const float* __restrict__ b1,
             const unsigned short* __restrict__ w2s, const float* __restrict__ b2,
             float* __restrict__ out, float* __restrict__ csum, float* __restrict__ csqs,
             int N) {
    __shared__ unsigned short sA[TILE_ROWS][LDS_STRIDE];  // bf16 activations
    __shared__ unsigned short sH[TILE_ROWS][LDS_STRIDE];  // bf16 hidden
    __shared__ float sSum[GIN_D];
    __shared__ float sSqs[GIN_D];

    const int tid  = threadIdx.x;
    const int wave = tid >> 5;
    const int lane = tid & 31;
    const int half = lane >> 4;
    const int ln   = lane & 15;
    const int row0 = blockIdx.x * TILE_ROWS;

    if (tid < GIN_D) { sSum[tid] = 0.f; sSqs[tid] = 0.f; }

    // cooperative load: 128x96 fp32 -> bf16 LDS (zero-fill past N)
#pragma unroll
    for (int it = 0; it < 12; ++it) {
        int idx = it * 256 + tid;          // float4 index in tile
        int r = idx / 24;
        int c = (idx % 24) * 4;
        int grow = row0 + r;
        v4f v = {0.f, 0.f, 0.f, 0.f};
        if (grow < N) v = *(const v4f*)(z + (size_t)grow * GIN_D + c);
        unsigned int p0 = (unsigned int)f2bf(v.x) | ((unsigned int)f2bf(v.y) << 16);
        unsigned int p1 = (unsigned int)f2bf(v.z) | ((unsigned int)f2bf(v.w) << 16);
        *(unsigned int*)&sA[r][c]     = p0;
        *(unsigned int*)&sA[r][c + 2] = p1;
    }
    __syncthreads();

    const int m0 = wave * 16;  // this wave's row tile within the block

    // ---- GEMM1: hidden = relu(A @ W1 + b1) ----
    v8f acc[6];
#pragma unroll
    for (int n = 0; n < 6; ++n) acc[n] = {};
#pragma unroll
    for (int kc = 0; kc < 3; ++kc) {
        FragU a;
        const unsigned short* ap = &sA[m0 + ln][kc * 32 + half * 16];
        a.q[0] = *(const v4u*)ap;
        a.q[1] = *(const v4u*)(ap + 8);
#pragma unroll
        for (int n = 0; n < 6; ++n) {
            FragU b;
            const v4u* wp = (const v4u*)w1s + (size_t)((n * 3 + kc) * 32 + lane) * 2;
            b.q[0] = wp[0];
            b.q[1] = wp[1];
            acc[n] = __builtin_amdgcn_wmma_f32_16x16x32_bf16(
                false, a.v, false, b.v, (short)0, acc[n], false, false);
        }
    }

    // bias + ReLU -> hidden tile (intra-wave region of sH)
#pragma unroll
    for (int n = 0; n < 6; ++n) {
        int col = n * 16 + ln;
        float bias = b1[col];
#pragma unroll
        for (int j = 0; j < 8; ++j) {
            float v = acc[n][j] + bias;
            v = fmaxf(v, 0.f);
            sH[m0 + half * 8 + j][col] = f2bf(v);
        }
    }
    __syncthreads();

    // ---- GEMM2: out = hidden @ W2 + b2 ----
#pragma unroll
    for (int n = 0; n < 6; ++n) acc[n] = {};
#pragma unroll
    for (int kc = 0; kc < 3; ++kc) {
        FragU a;
        const unsigned short* ap = &sH[m0 + ln][kc * 32 + half * 16];
        a.q[0] = *(const v4u*)ap;
        a.q[1] = *(const v4u*)(ap + 8);
#pragma unroll
        for (int n = 0; n < 6; ++n) {
            FragU b;
            const v4u* wp = (const v4u*)w2s + (size_t)((n * 3 + kc) * 32 + lane) * 2;
            b.q[0] = wp[0];
            b.q[1] = wp[1];
            acc[n] = __builtin_amdgcn_wmma_f32_16x16x32_bf16(
                false, a.v, false, b.v, (short)0, acc[n], false, false);
        }
    }

    // epilogue: store f32 output + per-channel sum / sum-of-squares for BN
#pragma unroll
    for (int n = 0; n < 6; ++n) {
        int col = n * 16 + ln;
        float bias = b2[col];
        float s = 0.f, s2 = 0.f;
#pragma unroll
        for (int j = 0; j < 8; ++j) {
            int grow = row0 + m0 + half * 8 + j;
            float v = acc[n][j] + bias;
            if (grow < N) {
                out[(size_t)grow * GIN_D + col] = v;
                s += v;
                s2 += v * v;
            }
        }
        atomicAdd(&sSum[col], s);
        atomicAdd(&sSqs[col], s2);
    }
    __syncthreads();
    if (tid < GIN_D)                 atomicAdd(&csum[tid], sSum[tid]);
    else if (tid < 2 * GIN_D)        atomicAdd(&csqs[tid - GIN_D], sSqs[tid - GIN_D]);
}

// ---------- kernel 5: BatchNorm (training-mode stats) + optional ReLU ----------
__global__ void gin_bn(const float* __restrict__ z, const float* __restrict__ csum,
                       const float* __restrict__ csqs, const float* __restrict__ gamma,
                       const float* __restrict__ beta, float* __restrict__ out,
                       int N, int do_relu) {
    int t = blockIdx.x * blockDim.x + threadIdx.x;
    int total = N * (GIN_D / 4);
    if (t >= total) return;
    int row = t / (GIN_D / 4);
    int c   = (t % (GIN_D / 4)) * 4;
    float invN = 1.0f / (float)N;
    v4f v = *(const v4f*)(z + (size_t)row * GIN_D + c);
    float r[4] = {v.x, v.y, v.z, v.w};
#pragma unroll
    for (int k = 0; k < 4; ++k) {
        float mean = csum[c + k] * invN;
        float var  = csqs[c + k] * invN - mean * mean;
        float f = (r[k] - mean) * __frsqrt_rn(var + 1e-5f) * gamma[c + k] + beta[c + k];
        if (do_relu) f = fmaxf(f, 0.f);
        r[k] = f;
    }
    v4f o = {r[0], r[1], r[2], r[3]};
    *(v4f*)(out + (size_t)row * GIN_D + c) = o;
}

// ---------- host orchestration ----------
extern "C" void kernel_launch(void* const* d_in, const int* in_sizes, int n_in,
                              void* d_out, int out_size, void* d_ws, size_t ws_size,
                              hipStream_t stream) {
    const int D = GIN_D;
    const float* x     = (const float*)d_in[0];
    const int*   ei    = (const int*)d_in[1];
    const float* W1    = (const float*)d_in[2];
    const float* b1    = (const float*)d_in[3];
    const float* W2    = (const float*)d_in[4];
    const float* b2    = (const float*)d_in[5];
    const float* gamma = (const float*)d_in[6];
    const float* beta  = (const float*)d_in[7];

    const int N = in_sizes[0] / D;
    const int E = in_sizes[1] / 2;
    const int* src = ei;
    const int* dst = ei + E;

    // workspace carve-up
    char* ws = (char*)d_ws;
    size_t nd = (size_t)N * D;
    float* zbuf = (float*)ws;  ws += nd * sizeof(float);
    float* obuf = (float*)ws;  ws += nd * sizeof(float);
    float* hbuf = (float*)ws;  ws += nd * sizeof(float);
    unsigned short* w1s = (unsigned short*)ws; ws += 9216 * sizeof(unsigned short);
    unsigned short* w2s = (unsigned short*)ws; ws += 9216 * sizeof(unsigned short);
    float* csum = (float*)ws;  ws += D * sizeof(float);
    float* csqs = (float*)ws;  ws += D * sizeof(float);

    const int n4 = (int)(nd / 4);
    const int copy_blocks    = (n4 + 255) / 256;
    const int scatter_tot    = E * 12;
    const int scatter_blocks = (scatter_tot + 255) / 256;
    const int prep_blocks    = (2 * 9216 + 255) / 256;
    const int mlp_blocks     = (N + TILE_ROWS - 1) / TILE_ROWS;
    const int bn_tot         = N * (D / 4);
    const int bn_blocks      = (bn_tot + 255) / 256;

    for (int i = 0; i < GIN_L; ++i) {
        const float* h = (i == 0) ? x : hbuf;
        float* bn_out = (i == GIN_L - 1) ? (float*)d_out : hbuf;

        gin_copy_f4<<<copy_blocks, 256, 0, stream>>>(h, zbuf, n4);
        gin_scatter<<<scatter_blocks, 256, 0, stream>>>(h, src, dst, zbuf, E);
        gin_prep<<<prep_blocks, 256, 0, stream>>>(W1 + (size_t)i * D * D,
                                                  W2 + (size_t)i * D * D,
                                                  w1s, w2s, csum, csqs);
        gin_mlp<<<mlp_blocks, 256, 0, stream>>>(zbuf, w1s, b1 + (size_t)i * D,
                                                w2s, b2 + (size_t)i * D,
                                                obuf, csum, csqs, N);
        gin_bn<<<bn_blocks, 256, 0, stream>>>(obuf, csum, csqs,
                                              gamma + (size_t)i * D, beta + (size_t)i * D,
                                              bn_out, N, (i < GIN_L - 1) ? 1 : 0);
    }
}
